// MOFVAE_83906481094957
// MI455X (gfx1250) — matrix operand order repaired and checked by reading, not compiled, not run on hardware
//
#include <hip/hip_runtime.h>
#include <hip/hip_bf16.h>

typedef __attribute__((ext_vector_type(16))) _Float16 v16h;
typedef __attribute__((ext_vector_type(8)))  _Float16 v8h;
typedef __attribute__((ext_vector_type(8)))  float    v8f;

#define WAVE 32
#define LDS_STRIDE 40     // halfs; 80B col stride: 16B-aligned fragments, conflict-free banks
#define LDS_HALFS 40960   // 80 KB: whole-W for (K/32)*F*40 <= 40960, else one 32xF slab

// ---------------- degree / normalization ----------------
__global__ void k_zero(float* p, int n) {
  int i = blockIdx.x * blockDim.x + threadIdx.x;
  if (i < n) p[i] = 0.0f;
}

__global__ void k_deg(const long long* __restrict__ ei, float* deg, int E) {
  int e = blockIdx.x * blockDim.x + threadIdx.x;
  if (e < E) {
    int d = (int)ei[(size_t)E + e];
    atomicAdd(deg + d, 1.0f);
  }
}

__global__ void k_dinv(float* deg, int n) {
  int i = blockIdx.x * blockDim.x + threadIdx.x;
  if (i < n) deg[i] = rsqrtf(deg[i] + 1.0f);
}

// ---------------- WMMA GEMM: g = (A@W)*dinv, agg = (A@W)*dinv^2 ----------------
// Block: 256 threads = 8 wave32. Block covers rowsPerBlock*16 rows x all F cols.
// Each wave computes a 16x64 strip = 4 accumulators sharing one A fragment.
// wholeW=1: all K/32 slabs of W staged once (80KB LDS), K-loop runs barrier-free.
// wholeW=0: one 32xF slab staged per K-step (layer 2: W too big for LDS).
__global__ void k_gemm_wmma(const float* __restrict__ A, const float* __restrict__ W,
                            const float* __restrict__ dinv,
                            float* __restrict__ g, float* __restrict__ agg,
                            int M, int K, int F, int logF, int wholeW) {
  __shared__ _Float16 ldsW[LDS_HALFS];

  const int tid  = threadIdx.x;
  const int lane = tid & (WAVE - 1);
  const int wv   = tid >> 5;
  const int wavesPerRow  = F >> 6;              // F/64: 1, 4, or 8
  const int rowsPerBlock = 8 / wavesPerRow;
  const int rowSub  = wv / wavesPerRow;
  const int colBase = (wv % wavesPerRow) << 6;  // 64-col strip per wave
  const int rowTile = blockIdx.x * rowsPerBlock + rowSub;
  const bool valid  = rowTile < (M >> 4);       // wave-uniform
  const int r  = lane & 15;
  const int hi = lane >> 4;                     // 0: lanes 0-15, 1: lanes 16-31
  const int row0 = rowTile << 4;
  const int ksel = hi << 3;
  const int slabHalfs = F * LDS_STRIDE;         // LDS halfs per 32-K slab

  // A fragment (16x32 f16): lanes 0-15 row r hold K {0..7,16..23}; lanes 16-31 K {8..15,24..31}
  const float* arow = A + (size_t)(row0 + r) * K + hi * 8;

  v8f acc[4] = {{}, {}, {}, {}};

  if (wholeW) {
    // stage ALL of W once: W[k][c] -> ldsW[(k/32)*slab + c*STRIDE + (k%32)]
    const int total = K << logF;
    for (int idx = tid; idx < total; idx += 256) {
      int c = idx & (F - 1);                    // fast index -> coalesced global read
      int k = idx >> logF;
      ldsW[(k >> 5) * slabHalfs + c * LDS_STRIDE + (k & 31)] = (_Float16)W[idx];
    }
    __syncthreads();

    if (valid) {                                // wave-uniform: EXEC all-1s for WMMA
      const _Float16* bbase = &ldsW[(colBase + r) * LDS_STRIDE + ksel];
      for (int k0 = 0; k0 < K; k0 += 32) {
        v16h a;
        const float* ap = arow + k0;
#pragma unroll
        for (int j = 0; j < 8; ++j) {
          a[j]     = (_Float16)ap[j];
          a[j + 8] = (_Float16)ap[16 + j];
        }
        // load all 4 B fragments first -> 4 independent back-to-back WMMAs
        v16h b[4];
#pragma unroll
        for (int ct = 0; ct < 4; ++ct) {
          const _Float16* bp = bbase + (k0 >> 5) * slabHalfs + ct * 16 * LDS_STRIDE;
          v8h blo = *(const v8h*)bp;
          v8h bhi = *(const v8h*)(bp + 16);
#pragma unroll
          for (int j = 0; j < 8; ++j) { b[ct][j] = blo[j]; b[ct][j + 8] = bhi[j]; }
        }
#pragma unroll
        for (int ct = 0; ct < 4; ++ct)
          acc[ct] = __builtin_amdgcn_wmma_f32_16x16x32_f16(false, a, false, b[ct],
                                                           (short)0, acc[ct], false, false);
      }
    }
  } else {
    // per-step staging (W too large for LDS)
    const int stageTotal = F << 5;              // 32*F elements per slab
    for (int k0 = 0; k0 < K; k0 += 32) {
      for (int idx = tid; idx < stageTotal; idx += 256) {
        int c  = idx & (F - 1);
        int kk = idx >> logF;
        ldsW[c * LDS_STRIDE + kk] = (_Float16)W[(size_t)(k0 + kk) * F + c];
      }
      __syncthreads();

      if (valid) {
        v16h a;
        const float* ap = arow + k0;
#pragma unroll
        for (int j = 0; j < 8; ++j) {
          a[j]     = (_Float16)ap[j];
          a[j + 8] = (_Float16)ap[16 + j];
        }
        v16h b[4];
#pragma unroll
        for (int ct = 0; ct < 4; ++ct) {
          const _Float16* bp = &ldsW[(colBase + ct * 16 + r) * LDS_STRIDE + ksel];
          v8h blo = *(const v8h*)bp;
          v8h bhi = *(const v8h*)(bp + 16);
#pragma unroll
          for (int j = 0; j < 8; ++j) { b[ct][j] = blo[j]; b[ct][j + 8] = bhi[j]; }
        }
#pragma unroll
        for (int ct = 0; ct < 4; ++ct)
          acc[ct] = __builtin_amdgcn_wmma_f32_16x16x32_f16(false, a, false, b[ct],
                                                           (short)0, acc[ct], false, false);
      }
      __syncthreads();                          // protect LDS before next stage
    }
  }

  if (valid) {
    // C/D layout: VGPR rr -> row rr (lanes 0-15) / rr+8 (lanes 16-31), col = (lane&15)
#pragma unroll
    for (int rr = 0; rr < 8; ++rr) {
      int row = row0 + rr + hi * 8;
      float dv = dinv[row];
      size_t ro = (size_t)row * F;
#pragma unroll
      for (int ct = 0; ct < 4; ++ct) {
        int col = colBase + ct * 16 + r;
        float v = acc[ct][rr];
        g[ro + col]   = v * dv;                 // g = hw * dinv
        agg[ro + col] = v * dv * dv;            // self-loop init: hw * dinv^2
      }
    }
  }
}

// ---------------- edge scatter-add: agg[dst] += g[src] * dinv[dst] ----------------
__global__ void k_edge_agg(const long long* __restrict__ ei,
                           const float* __restrict__ g,
                           const float* __restrict__ dinv,
                           float* __restrict__ agg, int E, int F) {
  const int lane = threadIdx.x & (WAVE - 1);
  const int e = blockIdx.x * (blockDim.x / WAVE) + (threadIdx.x / WAVE);
  if (e >= E) return;
  const int s = (int)ei[e];
  const int d = (int)ei[(size_t)E + e];
  const float w = dinv[d];
  const float4* gs = (const float4*)(g + (size_t)s * F);
  float* ad = agg + (size_t)d * F;
  for (int f4 = lane; f4 * 4 < F; f4 += WAVE) {
    float4 v = gs[f4];
    int f = f4 * 4;
    atomicAdd(ad + f + 0, v.x * w);
    atomicAdd(ad + f + 1, v.y * w);
    atomicAdd(ad + f + 2, v.z * w);
    atomicAdd(ad + f + 3, v.w * w);
  }
}

// ---------------- bias (+ optional relu) ----------------
__global__ void k_bias_act(const float* __restrict__ acc, const float* __restrict__ bias,
                           float* __restrict__ out, int total, int F, int relu) {
  int i = blockIdx.x * blockDim.x + threadIdx.x;
  if (i < total) {
    float v = acc[i] + bias[i & (F - 1)];   // F is a power of two
    out[i] = (relu && v < 0.0f) ? 0.0f : v;
  }
}

// ---------------- reparameterize ----------------
__global__ void k_z(const float* __restrict__ mu, const float* __restrict__ ls,
                    const float* __restrict__ eps, float* __restrict__ z, int total) {
  int i = blockIdx.x * blockDim.x + threadIdx.x;
  if (i < total) z[i] = mu[i] + eps[i] * __expf(ls[i]);
}

// ---------------- decoder: sigmoid(<z[src], z[dst]>) ----------------
__global__ void k_decode(const long long* __restrict__ ei, const float* __restrict__ z,
                         float* __restrict__ out, int E) {
  int e = blockIdx.x * blockDim.x + threadIdx.x;
  if (e < E) {
    int s = (int)ei[e];
    int d = (int)ei[(size_t)E + e];
    const float4* zs = (const float4*)(z + (size_t)s * 64);
    const float4* zd = (const float4*)(z + (size_t)d * 64);
    float acc = 0.0f;
#pragma unroll
    for (int j = 0; j < 16; ++j) {
      float4 a = zs[j], b = zd[j];
      acc += a.x * b.x + a.y * b.y + a.z * b.z + a.w * b.w;
    }
    out[e] = 1.0f / (1.0f + __expf(-acc));
  }
}

extern "C" void kernel_launch(void* const* d_in, const int* in_sizes, int n_in,
                              void* d_out, int out_size, void* d_ws, size_t ws_size,
                              hipStream_t stream) {
  const float*     x   = (const float*)d_in[0];
  const long long* ei  = (const long long*)d_in[1];   // int64 [2, E]
  const float*     eps = (const float*)d_in[2];
  const float*     W1  = (const float*)d_in[3];
  const float*     b1  = (const float*)d_in[4];
  const float*     W2  = (const float*)d_in[5];
  const float*     b2  = (const float*)d_in[6];
  const float*     Wmu = (const float*)d_in[7];
  const float*     bmu = (const float*)d_in[8];
  const float*     Wls = (const float*)d_in[9];
  const float*     bls = (const float*)d_in[10];
  float* out = (float*)d_out;

  const int IN = 128, H1 = 256, H2 = 512, L = 64;
  const int N = in_sizes[0] / IN;
  const int E = in_sizes[1] / 2;

  // workspace layout
  char* ws = (char*)d_ws;
  const size_t BUF = (size_t)N * H2 * sizeof(float);   // 102.4 MB at F=512
  float* dinv = (float*)ws;
  float* bufA = (float*)(ws + (1 << 20));              // g   (x@W scaled by dinv)
  float* bufB = (float*)(ws + (1 << 20) + BUF);        // agg accumulator
  float* bufC = (float*)(ws + (1 << 20) + 2 * BUF);    // layer activation

  float* out_mu = out + E;
  float* out_ls = out + E + (size_t)N * L;

  auto cdiv = [](long long a, long long b) { return (int)((a + b - 1) / b); };
  const int BT = 256;                                  // 8 wave32 per block
  const int rowTiles = N / 16;                         // 3125

  // symmetric normalization
  k_zero<<<cdiv(N, BT), BT, 0, stream>>>(dinv, N);
  k_deg <<<cdiv(E, BT), BT, 0, stream>>>(ei, dinv, E);
  k_dinv<<<cdiv(N, BT), BT, 0, stream>>>(dinv, N);

  // ---- GCN layer 1: x @ W1 -> 256, relu ----  (W1 f16: 4 slabs * 256 * 40 = 80KB -> whole)
  {
    int rpb = 8 / (H1 >> 6);                           // rowsPerBlock = 2
    k_gemm_wmma<<<cdiv(rowTiles, rpb), BT, 0, stream>>>(x, W1, dinv, bufA, bufB,
                                                        N, IN, H1, 8, 1);
    k_edge_agg <<<cdiv(E, 8), BT, 0, stream>>>(ei, bufA, dinv, bufB, E, H1);
    k_bias_act <<<cdiv((long long)N * H1, BT), BT, 0, stream>>>(bufB, b1, bufC,
                                                                N * H1, H1, 1);
  }
  // ---- GCN layer 2: h1 @ W2 -> 512, relu ----  (W2 f16 too big -> per-step staging)
  {
    int rpb = 8 / (H2 >> 6);                           // rowsPerBlock = 1
    k_gemm_wmma<<<cdiv(rowTiles, rpb), BT, 0, stream>>>(bufC, W2, dinv, bufA, bufB,
                                                        N, H1, H2, 9, 0);
    k_edge_agg <<<cdiv(E, 8), BT, 0, stream>>>(ei, bufA, dinv, bufB, E, H2);
    k_bias_act <<<cdiv((long long)N * H2, BT), BT, 0, stream>>>(bufB, b2, bufC,
                                                                N * H2, H2, 1);
  }
  // ---- mu head: h2 @ Wmu -> 64 ----  (16 slabs * 64 * 40 = 80KB -> whole)
  {
    int rpb = 8 / (L >> 6);                            // rowsPerBlock = 8
    k_gemm_wmma<<<cdiv(rowTiles, rpb), BT, 0, stream>>>(bufC, Wmu, dinv, bufA, bufB,
                                                        N, H2, L, 6, 1);
    k_edge_agg <<<cdiv(E, 8), BT, 0, stream>>>(ei, bufA, dinv, bufB, E, L);
    k_bias_act <<<cdiv((long long)N * L, BT), BT, 0, stream>>>(bufB, bmu, out_mu,
                                                               N * L, L, 0);
  }
  // ---- logstd head: h2 @ Wls -> 64 ----
  {
    int rpb = 8 / (L >> 6);
    k_gemm_wmma<<<cdiv(rowTiles, rpb), BT, 0, stream>>>(bufC, Wls, dinv, bufA, bufB,
                                                        N, H2, L, 6, 1);
    k_edge_agg <<<cdiv(E, 8), BT, 0, stream>>>(ei, bufA, dinv, bufB, E, L);
    k_bias_act <<<cdiv((long long)N * L, BT), BT, 0, stream>>>(bufB, bls, out_ls,
                                                               N * L, L, 0);
  }
  // ---- reparameterize + decode ----
  k_z     <<<cdiv((long long)N * L, BT), BT, 0, stream>>>(out_mu, out_ls, eps, bufA, N * L);
  k_decode<<<cdiv(E, BT), BT, 0, stream>>>(ei, bufA, out, E);
}
